// GATNet_27857157882203
// MI455X (gfx1250) — matrix operand order repaired and checked by reading, not compiled, not run on hardware
//
#include <hip/hip_runtime.h>
#include <hip/hip_bf16.h>

// ---------------------------------------------------------------------------
// GATv2 (2 layers) for MI455X / gfx1250.
//  - GEMMs via v_wmma_f32_16x16x32_bf16 (fp32 accumulate), one wave per 16x16
//    tile, K fully unrolled (K=128 -> 4 WMMAs), B pre-packed to fragment layout
//  - Edge phase: one wave32 per edge, float4 gathers, shfl_xor reductions,
//    f32 global atomics; segment max via monotonic-uint atomicMax
// ---------------------------------------------------------------------------

typedef __attribute__((ext_vector_type(16))) __bf16 v16bf;
typedef __attribute__((ext_vector_type(8)))  float  v8f;

union FragBF { v16bf v; unsigned short s[16]; unsigned int d[8]; uint4 q[2]; };
union FragF  { v8f  v; float f[8]; };

__device__ __forceinline__ unsigned enc_ord(float f) {
    unsigned u = __float_as_uint(f);
    return (u >> 31) ? ~u : (u | 0x80000000u);
}
__device__ __forceinline__ float dec_ord(unsigned u) {
    return (u & 0x80000000u) ? __uint_as_float(u & 0x7FFFFFFFu)
                             : __uint_as_float(~u);
}
__device__ __forceinline__ unsigned short f2bf(float f) {
    unsigned u = __float_as_uint(f);
    u += 0x7FFFu + ((u >> 16) & 1u);           // round-to-nearest-even
    return (unsigned short)(u >> 16);
}

// K offset inside a 32-wide k-tile for fragment register v (0..7), half h (0/1):
// (v<4?0:16) + h*8 + (v%4)*2
__device__ __host__ __forceinline__ int frag_kb(int v, int h) {
    return ((v & 4) << 2) + h * 8 + ((v & 3) << 1);
}

// ---------------- utility kernels ----------------
__global__ void fill_u32_kernel(unsigned int* p, unsigned int val, size_t n) {
    size_t i = (size_t)blockIdx.x * blockDim.x + threadIdx.x;
    if (i < n) p[i] = val;
}

__global__ void cvt_f32_bf16_kernel(const float* __restrict__ in,
                                    unsigned short* __restrict__ out, size_t n) {
    size_t i = (size_t)blockIdx.x * blockDim.x + threadIdx.x;
    if (i < n) out[i] = f2bf(in[i]);
}

// Pack W[K,N] (f32) directly into per-(ntile,ktile) WMMA B fragments (bf16).
// Fragment layout: Bp[((ntile*KT + ktile)*32 + lane)*16 + j], j = 2*v + p,
// so each lane's 16 bf16 are 32 contiguous bytes -> two b128 loads in the GEMM.
__global__ void pack_b_frag_kernel(const float* __restrict__ W,
                                   unsigned short* __restrict__ Bp,
                                   int K, int N) {
    const int KT = K / 32;
    size_t t = (size_t)blockIdx.x * blockDim.x + threadIdx.x;
    if (t >= (size_t)K * N) return;
    const int j     = (int)(t & 15);
    const int lane  = (int)((t >> 4) & 31);
    const int ktile = (int)((t >> 9) % KT);
    const int ntile = (int)((t >> 9) / KT);
    const int v = j >> 1, p = j & 1;
    const int k = ktile * 32 + frag_kb(v, lane >> 4) + p;
    const int n = ntile * 16 + (lane & 15);
    Bp[t] = f2bf(W[(size_t)k * N + n]);
}

// ---------------- WMMA GEMM: C[M,N] = A[M,K](bf16) * Bfrag + bias[N] ----------------
// One wave per 16x16 tile; 4 waves per block cover 4 consecutive N-tiles.
// Requires M % 16 == 0 (true here: M = 50000).
template <int K>
__global__ __launch_bounds__(128)
void gemm_bf16_wmma_kernel(const unsigned short* __restrict__ A,
                           const unsigned short* __restrict__ Bp,  // packed frags
                           const float* __restrict__ bias,
                           float* __restrict__ C,
                           int M, int N) {
    constexpr int KT = K / 32;
    const int lane  = threadIdx.x & 31;
    const int wave  = threadIdx.x >> 5;
    const int m0    = blockIdx.x * 16;
    const int n0    = (blockIdx.y * 4 + wave) * 16;
    if (n0 >= N) return;

    const int khalf = lane >> 4;     // 0: K 0..7 / 16..23   1: K 8..15 / 24..31
    const int l16   = lane & 15;

    FragF c;
#pragma unroll
    for (int i = 0; i < 8; ++i) c.f[i] = 0.0f;

    const unsigned short* arow = A + (size_t)(m0 + l16) * K + khalf * 8;
    const uint4* bfrag = (const uint4*)(Bp + ((size_t)((n0 >> 4) * KT) * 32 + lane) * 16);

#pragma unroll
    for (int kt = 0; kt < KT; ++kt) {
        FragBF a, b;
        // A: 8 bf16 at k0+khalf*8 and 8 bf16 at k0+16+khalf*8 -> two b128 loads
        a.q[0] = *(const uint4*)(arow + kt * 32);
        a.q[1] = *(const uint4*)(arow + kt * 32 + 16);
        // B: 32 contiguous bytes per lane -> two b128 loads
        b.q[0] = bfrag[64 * kt];
        b.q[1] = bfrag[64 * kt + 1];
        c.v = __builtin_amdgcn_wmma_f32_16x16x32_bf16(
                  false, a.v, false, b.v, (short)0, c.v, false, false);
    }

    const int   col   = n0 + l16;
    const float biasv = bias[col];
    float* crow = C + (size_t)(m0 + khalf * 8) * N + col;
#pragma unroll
    for (int r = 0; r < 8; ++r)
        crow[(size_t)r * N] = c.f[r] + biasv;
}

// ---------------- edge logits + segment max ----------------
// One wave per edge. H heads of C channels; 32/H lanes per head.
template <int H, int C>
__global__ void edge_logits_kernel(const float* __restrict__ xl,
                                   const float* __restrict__ xr,
                                   const long long* __restrict__ ei,
                                   int E0, int ET,
                                   const float* __restrict__ att,
                                   float* __restrict__ logits,
                                   unsigned int* __restrict__ lmax_enc) {
    constexpr int HC  = H * C;
    constexpr int LPH = 32 / H;      // lanes per head
    constexpr int CPL = HC / 32;     // channels per lane (4 for H=2, 2 for H=1)
    const int lane = threadIdx.x & 31;
    const int e    = blockIdx.x * (blockDim.x >> 5) + (threadIdx.x >> 5);
    if (e >= ET) return;

    int src, dst;
    if (e < E0) { src = (int)ei[e]; dst = (int)ei[E0 + e]; }
    else        { src = dst = e - E0; }

    const int head = lane / LPH;
    const int li   = lane % LPH;
    const int c0   = head * C + li * CPL;

    float xlv[CPL], xrv[CPL], atv[CPL];
    if constexpr (CPL == 4) {
        *(float4*)xlv = *(const float4*)(xl + (size_t)src * HC + c0);
        *(float4*)xrv = *(const float4*)(xr + (size_t)dst * HC + c0);
        *(float4*)atv = *(const float4*)(att + c0);
    } else {
        *(float2*)xlv = *(const float2*)(xl + (size_t)src * HC + c0);
        *(float2*)xrv = *(const float2*)(xr + (size_t)dst * HC + c0);
        *(float2*)atv = *(const float2*)(att + c0);
    }

    float s = 0.0f;
#pragma unroll
    for (int j = 0; j < CPL; ++j) {
        float m = xlv[j] + xrv[j];
        m = (m > 0.0f) ? m : 0.2f * m;          // GATv2 leaky_relu slope
        s += m * atv[j];
    }
#pragma unroll
    for (int msk = 1; msk < LPH; msk <<= 1) s += __shfl_xor(s, msk, 32);

    if (li == 0) {
        logits[(size_t)e * H + head] = s;
        atomicMax(&lmax_enc[(size_t)dst * H + head], enc_ord(s));
    }
}

// ---------------- exp + segment sum ----------------
__global__ void edge_exp_kernel(float* __restrict__ logits,   // in: logit, out: ex
                                const long long* __restrict__ ei,
                                int E0, int ET, int H,
                                const unsigned int* __restrict__ lmax_enc,
                                float* __restrict__ denom) {
    const size_t t = (size_t)blockIdx.x * blockDim.x + threadIdx.x;
    if (t >= (size_t)ET * H) return;
    const int e = (int)(t / H);
    const int h = (int)(t - (size_t)e * H);
    const int dst = (e < E0) ? (int)ei[E0 + e] : (e - E0);
    const float ex = __expf(logits[t] - dec_ord(lmax_enc[(size_t)dst * H + h]));
    logits[t] = ex;
    atomicAdd(&denom[(size_t)dst * H + h], ex);
}

// ---------------- weighted aggregation (scatter) ----------------
template <int H, int C>
__global__ void edge_aggregate_kernel(const float* __restrict__ xl,
                                      const float* __restrict__ ex,
                                      const float* __restrict__ denom,
                                      const long long* __restrict__ ei,
                                      int E0, int ET,
                                      float* __restrict__ agg) {
    constexpr int HC  = H * C;
    constexpr int LPH = 32 / H;
    constexpr int CPL = HC / 32;
    const int lane = threadIdx.x & 31;
    const int e    = blockIdx.x * (blockDim.x >> 5) + (threadIdx.x >> 5);
    if (e >= ET) return;

    int src, dst;
    if (e < E0) { src = (int)ei[e]; dst = (int)ei[E0 + e]; }
    else        { src = dst = e - E0; }

    const int head = lane / LPH;
    const int li   = lane % LPH;
    const int c0   = head * C + li * CPL;

    const float alpha = ex[(size_t)e * H + head] / denom[(size_t)dst * H + head];

    float xlv[CPL];
    if constexpr (CPL == 4)
        *(float4*)xlv = *(const float4*)(xl + (size_t)src * HC + c0);
    else
        *(float2*)xlv = *(const float2*)(xl + (size_t)src * HC + c0);

#pragma unroll
    for (int j = 0; j < CPL; ++j)
        atomicAdd(&agg[(size_t)dst * HC + c0 + j], alpha * xlv[j]);
}

// ---------------- inter-layer: +bias, leaky_relu(0.01), -> bf16 ----------------
__global__ void finish1_kernel(const float* __restrict__ agg,
                               const float* __restrict__ bias,
                               unsigned short* __restrict__ h_bf,
                               size_t n, int HC) {
    size_t i = (size_t)blockIdx.x * blockDim.x + threadIdx.x;
    if (i >= n) return;
    float v = agg[i] + bias[i % HC];
    v = (v > 0.0f) ? v : 0.01f * v;
    h_bf[i] = f2bf(v);
}

// ---------------- final: +bias2 -> d_out ----------------
__global__ void final_kernel(const float* __restrict__ agg,
                             const float* __restrict__ bias,
                             float* __restrict__ out,
                             size_t n, int C) {
    size_t i = (size_t)blockIdx.x * blockDim.x + threadIdx.x;
    if (i >= n) return;
    out[i] = agg[i] + bias[i % C];
}

// ---------------------------------------------------------------------------
extern "C" void kernel_launch(void* const* d_in, const int* in_sizes, int n_in,
                              void* d_out, int out_size, void* d_ws, size_t ws_size,
                              hipStream_t stream) {
    (void)n_in; (void)out_size; (void)ws_size;

    const float*     x     = (const float*)d_in[0];
    const long long* ei    = (const long long*)d_in[1];
    const float*     w1l   = (const float*)d_in[2];
    const float*     b1l   = (const float*)d_in[3];
    const float*     w1r   = (const float*)d_in[4];
    const float*     b1r   = (const float*)d_in[5];
    const float*     att1  = (const float*)d_in[6];
    const float*     bias1 = (const float*)d_in[7];
    const float*     w2l   = (const float*)d_in[8];
    const float*     b2l   = (const float*)d_in[9];
    const float*     w2r   = (const float*)d_in[10];
    const float*     b2r   = (const float*)d_in[11];
    const float*     att2  = (const float*)d_in[12];
    const float*     bias2 = (const float*)d_in[13];

    const int Fin = 128;                 // layer-1 input channels
    const int HC1 = 128;                 // heads*C1 = 2*64
    const int C2  = 64;                  // layer-2 out channels (H=1)
    const int Nn  = in_sizes[0] / Fin;   // 50000 nodes
    const int E0  = in_sizes[1] / 2;     // 800000 raw edges
    const int ET  = E0 + Nn;             // + self loops

    // ---- workspace carve-out ----
    char*  wsb = (char*)d_ws;
    size_t off = 0;
    auto alloc = [&](size_t bytes) -> void* {
        void* p = wsb + off;
        off = (off + bytes + 255) & ~(size_t)255;
        return p;
    };
    unsigned short* x_bf   = (unsigned short*)alloc((size_t)Nn * Fin * 2);
    unsigned short* w1l_bp = (unsigned short*)alloc((size_t)Fin * HC1 * 2);
    unsigned short* w1r_bp = (unsigned short*)alloc((size_t)Fin * HC1 * 2);
    unsigned short* w2l_bp = (unsigned short*)alloc((size_t)HC1 * C2 * 2);
    unsigned short* w2r_bp = (unsigned short*)alloc((size_t)HC1 * C2 * 2);
    float*          xl1    = (float*)alloc((size_t)Nn * HC1 * 4);
    float*          xr1    = (float*)alloc((size_t)Nn * HC1 * 4);
    float*          lg1    = (float*)alloc((size_t)ET * 2 * 4);   // logits -> ex
    unsigned int*   lmax1  = (unsigned int*)alloc((size_t)Nn * 2 * 4);
    float*          den1   = (float*)alloc((size_t)Nn * 2 * 4);
    float*          agg1   = (float*)alloc((size_t)Nn * HC1 * 4);
    unsigned short* h_bf   = (unsigned short*)alloc((size_t)Nn * HC1 * 2);
    float*          xl2    = (float*)alloc((size_t)Nn * C2 * 4);
    float*          xr2    = (float*)alloc((size_t)Nn * C2 * 4);
    float*          lg2    = (float*)alloc((size_t)ET * 4);
    unsigned int*   lmax2  = (unsigned int*)alloc((size_t)Nn * 4);
    float*          den2   = (float*)alloc((size_t)Nn * 4);
    float*          agg2   = (float*)alloc((size_t)Nn * C2 * 4);

    const int TB = 256;
    auto gr1  = [&](size_t n) { return (unsigned)((n + TB - 1) / TB); };
    const unsigned grEdge = (unsigned)((ET + 7) / 8);   // 8 waves per 256-thread block

    // ---- zero / init ----
    fill_u32_kernel<<<gr1((size_t)Nn * HC1), TB, 0, stream>>>((unsigned*)agg1, 0u, (size_t)Nn * HC1);
    fill_u32_kernel<<<gr1((size_t)Nn * 2), TB, 0, stream>>>(lmax1, 0u, (size_t)Nn * 2);
    fill_u32_kernel<<<gr1((size_t)Nn * 2), TB, 0, stream>>>((unsigned*)den1, 0u, (size_t)Nn * 2);
    fill_u32_kernel<<<gr1((size_t)Nn * C2), TB, 0, stream>>>((unsigned*)agg2, 0u, (size_t)Nn * C2);
    fill_u32_kernel<<<gr1((size_t)Nn), TB, 0, stream>>>(lmax2, 0u, (size_t)Nn);
    fill_u32_kernel<<<gr1((size_t)Nn), TB, 0, stream>>>((unsigned*)den2, 0u, (size_t)Nn);

    // ---- bf16 conversion (activations) + weight fragment packing ----
    cvt_f32_bf16_kernel<<<gr1((size_t)Nn * Fin), TB, 0, stream>>>(x, x_bf, (size_t)Nn * Fin);
    pack_b_frag_kernel<<<gr1((size_t)Fin * HC1), TB, 0, stream>>>(w1l, w1l_bp, Fin, HC1);
    pack_b_frag_kernel<<<gr1((size_t)Fin * HC1), TB, 0, stream>>>(w1r, w1r_bp, Fin, HC1);
    pack_b_frag_kernel<<<gr1((size_t)HC1 * C2), TB, 0, stream>>>(w2l, w2l_bp, HC1, C2);
    pack_b_frag_kernel<<<gr1((size_t)HC1 * C2), TB, 0, stream>>>(w2r, w2r_bp, HC1, C2);

    // ---- layer 1 GEMMs: [Nn,128]x[128,128] ----
    {
        dim3 g(Nn / 16, (HC1 / 16 + 3) / 4);
        gemm_bf16_wmma_kernel<128><<<g, 128, 0, stream>>>(x_bf, w1l_bp, b1l, xl1, Nn, HC1);
        gemm_bf16_wmma_kernel<128><<<g, 128, 0, stream>>>(x_bf, w1r_bp, b1r, xr1, Nn, HC1);
    }

    // ---- layer 1 edge phase (H=2, C=64) ----
    edge_logits_kernel<2, 64><<<grEdge, TB, 0, stream>>>(xl1, xr1, ei, E0, ET, att1, lg1, lmax1);
    edge_exp_kernel<<<gr1((size_t)ET * 2), TB, 0, stream>>>(lg1, ei, E0, ET, 2, lmax1, den1);
    edge_aggregate_kernel<2, 64><<<grEdge, TB, 0, stream>>>(xl1, lg1, den1, ei, E0, ET, agg1);

    // ---- inter-layer activation -> bf16 ----
    finish1_kernel<<<gr1((size_t)Nn * HC1), TB, 0, stream>>>(agg1, bias1, h_bf, (size_t)Nn * HC1, HC1);

    // ---- layer 2 GEMMs: [Nn,128]x[128,64] ----
    {
        dim3 g(Nn / 16, (C2 / 16 + 3) / 4);
        gemm_bf16_wmma_kernel<128><<<g, 128, 0, stream>>>(h_bf, w2l_bp, b2l, xl2, Nn, C2);
        gemm_bf16_wmma_kernel<128><<<g, 128, 0, stream>>>(h_bf, w2r_bp, b2r, xr2, Nn, C2);
    }

    // ---- layer 2 edge phase (H=1, C=64) ----
    edge_logits_kernel<1, 64><<<grEdge, TB, 0, stream>>>(xl2, xr2, ei, E0, ET, att2, lg2, lmax2);
    edge_exp_kernel<<<gr1((size_t)ET), TB, 0, stream>>>(lg2, ei, E0, ET, 1, lmax2, den2);
    edge_aggregate_kernel<1, 64><<<grEdge, TB, 0, stream>>>(xl2, lg2, den2, ei, E0, ET, agg2);

    // ---- final bias add -> d_out [Nn,64] fp32 ----
    final_kernel<<<gr1((size_t)Nn * C2), TB, 0, stream>>>(agg2, bias2, (float*)d_out,
                                                          (size_t)Nn * C2, C2);
}
